// GNN_18373870092569
// MI455X (gfx1250) — matrix operand order repaired and checked by reading, not compiled, not run on hardware
//
#include <hip/hip_runtime.h>
#include <math.h>

// ---------------------------------------------------------------------------
// TAGConv GNN for MI455X (gfx1250, wave32).
//  * Sparse propagation: L2-resident gather/scatter with global_atomic_add_f32
//  * Dense per-hop transforms: V_WMMA_F32_16X16X4_F32 (fp32, matches reference)
//  * N = 200000 is a multiple of 16 -> full WMMA tiles, EXEC all-ones.
// ---------------------------------------------------------------------------

typedef float    v2f  __attribute__((ext_vector_type(2)));
typedef float    v8f  __attribute__((ext_vector_type(8)));
typedef _Float16 v16h __attribute__((ext_vector_type(16)));

// ---------------------------------- utility --------------------------------

__global__ void zero_kernel(float* __restrict__ p, long long n) {
  long long i = (long long)blockIdx.x * blockDim.x + threadIdx.x;
  if (i < n) p[i] = 0.0f;
}

__global__ void deg_kernel(const int* __restrict__ col, float* __restrict__ deg, int E) {
  int e = blockIdx.x * blockDim.x + threadIdx.x;
  if (e < E) atomicAdd(&deg[col[e]], 1.0f);
}

__global__ void dinv_kernel(float* __restrict__ deg, int N) {
  int i = blockIdx.x * blockDim.x + threadIdx.x;
  if (i < N) {
    float d = deg[i];
    deg[i] = (d > 0.0f) ? rsqrtf(fmaxf(d, 1.0f)) : 0.0f;
  }
}

__global__ void norm_kernel(const int* __restrict__ row, const int* __restrict__ col,
                            const float* __restrict__ dinv, float* __restrict__ nrm, int E) {
  int e = blockIdx.x * blockDim.x + threadIdx.x;
  if (e < E) nrm[e] = dinv[row[e]] * dinv[col[e]];
}

__global__ void relu_kernel(const float* __restrict__ in, float* __restrict__ out, int n) {
  int i = blockIdx.x * blockDim.x + threadIdx.x;
  if (i < n) out[i] = fmaxf(in[i], 0.0f);
}

// acc[i] = sum_k b[k][i % C]   (4 hop biases summed once)
template<int C>
__global__ void bias_init_kernel(float* __restrict__ acc, const float* __restrict__ b, int NC) {
  int i = blockIdx.x * blockDim.x + threadIdx.x;
  if (i >= NC) return;
  int c = i & (C - 1);
  float s = 0.0f;
#pragma unroll
  for (int k = 0; k < 4; ++k) s += b[k * C + c];
  acc[i] = s;
}

// ------------------------- sparse propagation (scatter) --------------------
// out[col[e]][:] += h[row[e]][:] * norm[e]; one thread per (edge, float4 chunk)
template<int C>
__global__ void propagate_kernel(const float* __restrict__ h, const int* __restrict__ row,
                                 const int* __restrict__ col, const float* __restrict__ nrm,
                                 float* __restrict__ out, int E) {
  constexpr int CH = C / 4;
  int t = blockIdx.x * blockDim.x + threadIdx.x;
  if (t >= E * CH) return;
  int e = t / CH;           // CH is a power of two -> shift
  int j = t - e * CH;
  int r = row[e], c = col[e];
  float w = nrm[e];
  float4 v = reinterpret_cast<const float4*>(h + (size_t)r * C)[j];
  float* dst = out + (size_t)c * C + j * 4;
  atomicAdd(dst + 0, v.x * w);
  atomicAdd(dst + 1, v.y * w);
  atomicAdd(dst + 2, v.z * w);
  atomicAdd(dst + 3, v.w * w);
}

// ------------------------------- WMMA GEMM ---------------------------------
// acc[M,Cout] += A[M,Cin] @ W[Cin,Cout]; one wave per 16x16 output tile.
// WMMA f32 layouts (ISA 7.12.2):
//   A 16x4:  lane(0-15) M=lane, VGPR{0,1}=K{0,1}; lanes 16-31: K{2,3}
//   B 4x16:  lane%16 = N, lanes 0-15: K{0,1}, lanes 16-31: K{2,3}
//   C 16x16: VGPR r: M = r + (lane/16)*8, N = lane%16
template<int Cin, int Cout>
__global__ void gemm_tile_kernel(const float* __restrict__ A, const float* __restrict__ W,
                                 float* __restrict__ acc, int tiles) {
  int gid  = blockIdx.x * blockDim.x + threadIdx.x;
  int wave = gid >> 5;
  if (wave >= tiles) return;                  // wave-uniform exit (EXEC stays full)
  int lane = threadIdx.x & 31;
  constexpr int tilesN = Cout / 16;
  int mt = wave / tilesN;
  int nt = wave - mt * tilesN;
  int lo = lane & 15;
  int hi = lane >> 4;
  int n  = nt * 16 + lo;
  int m0 = mt * 16;

#if defined(__HIP_DEVICE_COMPILE__) && __has_builtin(__builtin_amdgcn_wmma_f32_16x16x4_f32)
  v8f c;
#pragma unroll
  for (int r = 0; r < 8; ++r)
    c[r] = acc[(size_t)(m0 + r + hi * 8) * Cout + n];
  int m = m0 + lo;
#pragma unroll
  for (int kk = 0; kk < Cin; kk += 4) {
    int k0 = kk + hi * 2;
    v2f a, b;
    a.x = A[(size_t)m * Cin + k0];
    a.y = A[(size_t)m * Cin + k0 + 1];
    b.x = W[(size_t)k0 * Cout + n];
    b.y = W[(size_t)(k0 + 1) * Cout + n];
    // 8 args: (neg_a, A, neg_b, B, c_mod, C, reuse_a, reuse_b)
    c = __builtin_amdgcn_wmma_f32_16x16x4_f32(false, a, false, b, (short)0, c, false, false);
  }
#pragma unroll
  for (int r = 0; r < 8; ++r)
    acc[(size_t)(m0 + r + hi * 8) * Cout + n] = c[r];

#elif defined(__HIP_DEVICE_COMPILE__) && __has_builtin(__builtin_amdgcn_wmma_f32_16x16x32_f16)
  // Fallback: codegen-confirmed f16 WMMA (K zero-padded when Cin==16).
  v8f c;
#pragma unroll
  for (int r = 0; r < 8; ++r)
    c[r] = acc[(size_t)(m0 + r + hi * 8) * Cout + n];
  int m = m0 + lo;
  v16h av, bv;
#pragma unroll
  for (int j = 0; j < 16; ++j) {
    int v = j >> 1, o = j & 1;
    int ka = ((v < 4) ? (v * 2) : (16 + (v - 4) * 2)) + hi * 8 + o; // A 16-bit layout
    int kb = ((v < 4) ? (v * 2) : (16 + (v - 4) * 2)) + hi * 8 + o; // B 16-bit layout
    av[j] = (ka < Cin) ? (_Float16)A[(size_t)m * Cin + ka] : (_Float16)0.0f;
    bv[j] = (kb < Cin) ? (_Float16)W[(size_t)kb * Cout + n] : (_Float16)0.0f;
  }
  c = __builtin_amdgcn_wmma_f32_16x16x32_f16(false, av, false, bv, (short)0, c, false, false);
#pragma unroll
  for (int r = 0; r < 8; ++r)
    acc[(size_t)(m0 + r + hi * 8) * Cout + n] = c[r];

#else
  // Scalar fallback (host-parse path; never the intended device path).
  for (int r = 0; r < 8; ++r) {
    int rowi = m0 + r + hi * 8;
    float s = acc[(size_t)rowi * Cout + n];
    for (int k = 0; k < Cin; ++k)
      s += A[(size_t)rowi * Cin + k] * W[(size_t)k * Cout + n];
    acc[(size_t)rowi * Cout + n] = s;
  }
#endif
}

// ------------------------------ pooling + head -----------------------------

__global__ void pool_kernel(const float* __restrict__ h, const int* __restrict__ batch,
                            float* __restrict__ sums, float* __restrict__ cnt, int N) {
  int t = blockIdx.x * blockDim.x + threadIdx.x;
  if (t >= N * 16) return;
  int i = t >> 4, c = t & 15;
  int g = batch[i];
  atomicAdd(&sums[g * 16 + c], h[t]);
  if (c == 0) atomicAdd(&cnt[g], 1.0f);
}

__global__ void head_kernel(const float* __restrict__ psum, const float* __restrict__ pcnt,
                            const float* __restrict__ Wfc, const float* __restrict__ bfc,
                            float* __restrict__ out) {
  int g = blockIdx.x;        // one 32-thread wave per graph
  int lane = threadIdx.x;
  __shared__ float pooled[16];
  __shared__ float logits[16];
  float c = fmaxf(pcnt[g], 1.0f);
  if (lane < 16) pooled[lane] = psum[g * 16 + lane] / c;
  __syncthreads();
  if (lane < 13) {
    float s = bfc[lane];
#pragma unroll
    for (int k = 0; k < 16; ++k) s += pooled[k] * Wfc[k * 13 + lane];
    logits[lane] = s;
  }
  __syncthreads();
  if (lane < 13) {
    float mx = logits[0];
    for (int j = 1; j < 13; ++j) mx = fmaxf(mx, logits[j]);
    float se = 0.0f;
    for (int j = 0; j < 13; ++j) se += expf(logits[j] - mx);
    out[g * 13 + lane] = (logits[lane] - mx) - logf(se);
  }
}

// --------------------------------- launch ----------------------------------

static inline int nblk(long long n, int t) { return (int)((n + t - 1) / t); }

extern "C" void kernel_launch(void* const* d_in, const int* in_sizes, int n_in,
                              void* d_out, int out_size, void* d_ws, size_t ws_size,
                              hipStream_t stream) {
  const float* x    = (const float*)d_in[0];
  const int*   eidx = (const int*)  d_in[1];
  const int*   batch= (const int*)  d_in[2];
  const float* W1   = (const float*)d_in[3];
  const float* b1   = (const float*)d_in[4];
  const float* W2   = (const float*)d_in[5];
  const float* b2   = (const float*)d_in[6];
  const float* W3   = (const float*)d_in[7];
  const float* b3   = (const float*)d_in[8];
  const float* Wfc  = (const float*)d_in[9];
  const float* bfc  = (const float*)d_in[10];
  float*       out  = (float*)d_out;

  const int N = in_sizes[0] / 32;   // 200000 (multiple of 16)
  const int E = in_sizes[1] / 2;    // 6400000
  const int G = out_size / 13;      // 64
  const int* row = eidx;
  const int* col = eidx + E;

  // workspace layout (floats)
  float* dinv = (float*)d_ws;                  // N      (deg -> d^{-1/2})
  float* nrm  = dinv + N;                      // E
  float* bufA = nrm + E;                       // N*32
  float* bufB = bufA + (size_t)N * 32;         // N*32
  float* acc  = bufB + (size_t)N * 32;         // N*32
  float* psum = acc  + (size_t)N * 32;         // G*16
  float* pcnt = psum + (size_t)G * 16;         // G
  (void)ws_size; (void)n_in;

  const int T = 256;

  // ---- gcn_norm ----
  zero_kernel<<<nblk(N, T), T, 0, stream>>>(dinv, N);
  deg_kernel<<<nblk(E, T), T, 0, stream>>>(col, dinv, E);
  dinv_kernel<<<nblk(N, T), T, 0, stream>>>(dinv, N);
  norm_kernel<<<nblk(E, T), T, 0, stream>>>(row, col, dinv, nrm, E);

  // ---- one TAGConv layer ----
  auto layer = [&](const float* input, const float* W, const float* b,
                   auto cinTag, auto coutTag) {
    constexpr int Cin  = decltype(cinTag)::value;
    constexpr int Cout = decltype(coutTag)::value;
    const int tiles = (N / 16) * (Cout / 16);
    bias_init_kernel<Cout><<<nblk((long long)N * Cout, T), T, 0, stream>>>(acc, b, N * Cout);
    gemm_tile_kernel<Cin, Cout><<<nblk((long long)tiles * 32, T), T, 0, stream>>>(input, W, acc, tiles);
    const float* h = input;
    float* ping[3] = {bufA, bufB, bufA};
    for (int k = 1; k <= 3; ++k) {
      float* nxt = ping[k - 1];
      zero_kernel<<<nblk((long long)N * Cin, T), T, 0, stream>>>(nxt, (long long)N * Cin);
      propagate_kernel<Cin><<<nblk((long long)E * (Cin / 4), T), T, 0, stream>>>(h, row, col, nrm, nxt, E);
      gemm_tile_kernel<Cin, Cout><<<nblk((long long)tiles * 32, T), T, 0, stream>>>(
          nxt, W + (size_t)k * Cin * Cout, acc, tiles);
      h = nxt;
    }
    relu_kernel<<<nblk((long long)N * Cout, T), T, 0, stream>>>(acc, bufB, N * Cout);
  };

  layer(x,    W1, b1, std::integral_constant<int, 32>{}, std::integral_constant<int, 16>{});
  layer(bufB, W2, b2, std::integral_constant<int, 16>{}, std::integral_constant<int, 32>{});
  layer(bufB, W3, b3, std::integral_constant<int, 32>{}, std::integral_constant<int, 16>{});

  // ---- global mean pool + FC + log_softmax ----
  zero_kernel<<<nblk((long long)G * 17, T), T, 0, stream>>>(psum, (long long)G * 17);
  pool_kernel<<<nblk((long long)N * 16, T), T, 0, stream>>>(bufB, batch, psum, pcnt, N);
  head_kernel<<<G, 32, 0, stream>>>(psum, pcnt, Wfc, bfc, out);
}
